// MultiHeadSelfAttention_33990371180567
// MI455X (gfx1250) — matrix operand order repaired and checked by reading, not compiled, not run on hardware
//
#include <hip/hip_runtime.h>

typedef __bf16 bf16;
typedef __bf16 bf16x8  __attribute__((ext_vector_type(8)));
typedef __bf16 bf16x16 __attribute__((ext_vector_type(16)));
typedef float  f32x8   __attribute__((ext_vector_type(8)));
typedef unsigned v4u   __attribute__((ext_vector_type(4)));
typedef int      v4i   __attribute__((ext_vector_type(4)));
typedef int      v8i   __attribute__((ext_vector_type(8)));

#define B_  4
#define S_  2048
#define D_  1024
#define H_  16
#define HD_ 64
#define M_TOT (B_ * S_)   // 8192

#if __has_builtin(__builtin_amdgcn_tensor_load_to_lds)
#define HAVE_TDM 1
#endif

__device__ __forceinline__ bf16 f2bf(float f) {
    union { float f; unsigned u; } c{f};
    unsigned r = (c.u + 0x7FFFu + ((c.u >> 16) & 1u)) >> 16;
    unsigned short us = (unsigned short)r;
    return __builtin_bit_cast(bf16, us);
}

__device__ __forceinline__ bf16 f2bf_fast(float a) {
#if __has_builtin(__builtin_amdgcn_cvt_pk_bf16_f32)
    auto v = __builtin_amdgcn_cvt_pk_bf16_f32(a, 0.0f);
    static_assert(sizeof(v) == 4, "cvt_pk_bf16_f32 result size");
    bf16 out;
    __builtin_memcpy(&out, &v, 2);
    return out;
#else
    return f2bf(a);
#endif
}

__device__ __forceinline__ void store_bf2(bf16* p, float a, float b) {
#if __has_builtin(__builtin_amdgcn_cvt_pk_bf16_f32)
    auto v = __builtin_amdgcn_cvt_pk_bf16_f32(a, b);
    __builtin_memcpy(p, &v, 4);
#else
    p[0] = f2bf(a); p[1] = f2bf(b);
#endif
}

__device__ __forceinline__ bf16x16 cat(bf16x8 lo, bf16x8 hi) {
    return __builtin_shufflevector(lo, hi, 0,1,2,3,4,5,6,7,8,9,10,11,12,13,14,15);
}

__device__ __forceinline__ f32x8 wmma_bf16(bf16x16 a, bf16x16 b, f32x8 c) {
    return __builtin_amdgcn_wmma_f32_16x16x32_bf16(false, a, false, b, (short)0, c, false, false);
}

__device__ __forceinline__ void wait_tensorcnt0() {
#if __has_builtin(__builtin_amdgcn_s_wait_tensorcnt)
    __builtin_amdgcn_s_wait_tensorcnt(0);
#else
    asm volatile("s_wait_tensorcnt 0x0" ::: "memory");
#endif
}
__device__ __forceinline__ void wait_tensorcnt1() {
#if __has_builtin(__builtin_amdgcn_s_wait_tensorcnt)
    __builtin_amdgcn_s_wait_tensorcnt(1);
#else
    asm volatile("s_wait_tensorcnt 0x1" ::: "memory");
#endif
}

// Compiler-level fence: LDS written behind the compiler's back by the TDM.
// Escaping the LDS base into a memory-clobbering asm keeps the fragment
// ds_loads from being folded to undef (they were, in round 3).
__device__ __forceinline__ void lds_acquire(const void* p) {
    asm volatile("" :: "r"(p) : "memory");
}

#ifdef HAVE_TDM
// 2-D TDM load: tile_d1 rows of tile_d0 contiguous bf16, row pitch stride0
// (elements).  pad_enable inserts 4 DWORDs (8 bf16) after every
// 2^(pad_iv+1) DWORDs, matching the padded LDS rows (32->40, 64->72 bf16).
__device__ __forceinline__ void tdm_load_2d(unsigned lds_addr, const void* gaddr,
                                            unsigned tensor_d0, unsigned tensor_d1,
                                            unsigned tile_d0, unsigned tile_d1,
                                            unsigned stride0, unsigned pad_iv) {
    unsigned long long ga = (unsigned long long)(size_t)gaddr;
    v4u g0;
    g0[0] = 1u;                                            // count=1 (valid D#)
    g0[1] = lds_addr;                                      // LDS byte address
    g0[2] = (unsigned)(ga & 0xFFFFFFFFull);                // global addr [31:0]
    g0[3] = (unsigned)((ga >> 32) & 0x1FFFFFFull) | (2u << 30);  // [56:32] | type=2
    v8i g1;
    g1[0] = (int)((1u << 16)          // data_size = 2 bytes
                | (1u << 20)          // pad_enable
                | (pad_iv << 22)      // pad_interval
                | (3u << 25));        // pad_amount: 4 DWORDs (8 bf16)
    g1[1] = (int)((tensor_d0 & 0xFFFFu) << 16);
    g1[2] = (int)(((tensor_d0 >> 16) & 0xFFFFu) | ((tensor_d1 & 0xFFFFu) << 16));
    g1[3] = (int)(((tensor_d1 >> 16) & 0xFFFFu) | ((tile_d0 & 0xFFFFu) << 16));
    g1[4] = (int)(tile_d1 & 0xFFFFu);
    g1[5] = (int)stride0;
    g1[6] = 0;
    g1[7] = 0;
    v4i z4 = {0, 0, 0, 0};
#if __clang_major__ >= 23
    v8i z8 = {0, 0, 0, 0, 0, 0, 0, 0};
    __builtin_amdgcn_tensor_load_to_lds(g0, g1, z4, z4, z8, 0);
#else
    __builtin_amdgcn_tensor_load_to_lds(g0, g1, z4, z4, 0);
#endif
}
#endif

// ---------------------------------------------------------------------------
// Kernel 0a: X fp32 -> bf16 (streaming, packed converts, b128 stores)
// ---------------------------------------------------------------------------
__global__ __launch_bounds__(256) void convert_x(const float* __restrict__ X,
                                                 bf16* __restrict__ Xb) {
    size_t i = ((size_t)blockIdx.x * 256 + threadIdx.x) * 8;
    float4 a = *(const float4*)&X[i];
    float4 b = *(const float4*)&X[i + 4];
    __align__(16) bf16 t[8];
    store_bf2(&t[0], a.x, a.y); store_bf2(&t[2], a.z, a.w);
    store_bf2(&t[4], b.x, b.y); store_bf2(&t[6], b.z, b.w);
    *(bf16x8*)&Xb[i] = *(const bf16x8*)t;
}

// ---------------------------------------------------------------------------
// Kernel 0b: weights fp32 [k][n] -> bf16 TRANSPOSED [n][k] (LDS-tiled).
// ---------------------------------------------------------------------------
__global__ __launch_bounds__(256) void convert_wt(
    const float* __restrict__ Wq, const float* __restrict__ Wk,
    const float* __restrict__ Wv, const float* __restrict__ Wo,
    bf16* __restrict__ Tq, bf16* __restrict__ Tk,
    bf16* __restrict__ Tv, bf16* __restrict__ To) {

    __shared__ bf16 t[64][72];
    const float* W;  bf16* T;
    if (blockIdx.z == 0)      { W = Wq; T = Tq; }
    else if (blockIdx.z == 1) { W = Wk; T = Tk; }
    else if (blockIdx.z == 2) { W = Wv; T = Tv; }
    else                      { W = Wo; T = To; }

    const int tid = threadIdx.x;
    const int k0 = blockIdx.y * 64, n0 = blockIdx.x * 64;
    #pragma unroll
    for (int p = 0; p < 4; ++p) {
        int row = p * 16 + (tid >> 4);
        int col = (tid & 15) * 4;
        float4 v = *(const float4*)&W[(size_t)(k0 + row) * D_ + n0 + col];
        store_bf2(&t[row][col], v.x, v.y);
        store_bf2(&t[row][col + 2], v.z, v.w);
    }
    __syncthreads();
    #pragma unroll
    for (int p = 0; p < 2; ++p) {
        int n = p * 32 + (tid >> 3);
        int kcol = (tid & 7) * 8;
        bf16x8 v;
        #pragma unroll
        for (int j = 0; j < 8; ++j) v[j] = t[kcol + j][n];
        *(bf16x8*)&T[(size_t)(n0 + n) * D_ + k0 + kcol] = v;
    }
}

// ---------------------------------------------------------------------------
// Shared GEMM core: C = A(bf16 [M][1024]) @ Wt^T (Wt bf16 [n][k]).
// 128x128 tile per 256-thread block; double-buffered TDM staging, symmetric:
// every wave issues one 32-row quarter-tile D# (waves 0-3 -> A, 4-7 -> B).
// mode 0: bf16 out [B,H,S,HD]; mode 1: bf16 out [B,H,HD,S]; mode 2: fp32 flat.
// ---------------------------------------------------------------------------
struct GemmSmem {
    bf16 Al[2][128][40];   // 10240 B each buffer
    bf16 Bl[2][128][40];
};
#define AL_OFF(buf) ((unsigned)(buf) * 10240u)
#define BL_OFF(buf) (20480u + (unsigned)(buf) * 10240u)

__device__ __forceinline__ void gemm_stage(GemmSmem& sm, int buf,
                                           const bf16* A, const bf16* Bt,
                                           int m0, int n0, int kk,
                                           int tid, int wave) {
#ifdef HAVE_TDM
    const bool isA = wave < 4;
    const bf16* src = isA ? A : Bt;
    int row0 = (isA ? m0 : n0) + (wave & 3) * 32;
    unsigned lds = (isA ? AL_OFF(buf) : BL_OFF(buf)) + (unsigned)(wave & 3) * 32u * 80u;
    tdm_load_2d(lds, &src[(size_t)row0 * D_ + kk], D_, M_TOT, 32, 32, D_, 3);
#else
    #pragma unroll
    for (int p = 0; p < 2; ++p) {
        int v = p * 256 + tid;
        int row = v >> 2, col = (v & 3) * 8;
        *(bf16x8*)&sm.Al[buf][row][col] =
            *(const bf16x8*)&A[(size_t)(m0 + row) * D_ + kk + col];
        *(bf16x8*)&sm.Bl[buf][row][col] =
            *(const bf16x8*)&Bt[(size_t)(n0 + row) * D_ + kk + col];
    }
#endif
}

__device__ __forceinline__ void gemm_core(GemmSmem& sm, const bf16* __restrict__ A,
                                          const bf16* __restrict__ Bt,
                                          void* OutP, int mode) {
    const int tid  = threadIdx.x;
    const int m0   = blockIdx.y * 128;
    const int n0   = blockIdx.x * 128;
    const int wave = tid >> 5, lane = tid & 31, half = lane >> 4, r = lane & 15;
    const int wm   = (wave & 3) * 32, wn = (wave >> 2) * 64;

    f32x8 acc[2][4] = {};
    const int T = D_ / 32;               // 32 k-tiles

    gemm_stage(sm, 0, A, Bt, m0, n0, 0, tid, wave);   // prologue
    for (int t = 0; t < T; ++t) {
        if (t + 1 < T) {
            gemm_stage(sm, (t + 1) & 1, A, Bt, m0, n0, (t + 1) * 32, tid, wave);
#ifdef HAVE_TDM
            wait_tensorcnt1();           // in-order: tile t has landed
#endif
        } else {
#ifdef HAVE_TDM
            wait_tensorcnt0();
#endif
        }
#ifdef HAVE_TDM
        lds_acquire(&sm);
#endif
        __syncthreads();
        const int buf = t & 1;
        bf16x16 a[2], b[4];
        #pragma unroll
        for (int ms = 0; ms < 2; ++ms) {
            int row = wm + ms * 16 + r;
            a[ms] = cat(*(const bf16x8*)&sm.Al[buf][row][half * 8],
                        *(const bf16x8*)&sm.Al[buf][row][16 + half * 8]);
        }
        #pragma unroll
        for (int ns = 0; ns < 4; ++ns) {
            int cn = wn + ns * 16 + r;
        b[ns] = cat(*(const bf16x8*)&sm.Bl[buf][cn][half * 8],
                        *(const bf16x8*)&sm.Bl[buf][cn][16 + half * 8]);
        }
        #pragma unroll
        for (int ms = 0; ms < 2; ++ms)
            #pragma unroll
            for (int ns = 0; ns < 4; ++ns)
                acc[ms][ns] = wmma_bf16(a[ms], b[ns], acc[ms][ns]);
        __syncthreads();
    }

    // epilogue: hoisted base pointers, constant per-i strides
    #pragma unroll
    for (int ms = 0; ms < 2; ++ms)
        #pragma unroll
        for (int ns = 0; ns < 4; ++ns) {
            int gm0 = m0 + wm + ms * 16 + half * 8;          // rows gm0..gm0+7
            int gn  = n0 + wn + ns * 16 + r;
            if (mode == 2) {
                float* p = (float*)OutP + (size_t)gm0 * D_ + gn;
                #pragma unroll
                for (int i = 0; i < 8; ++i) p[(size_t)i * D_] = acc[ms][ns][i];
            } else {
                int bb = gm0 >> 11, s0 = gm0 & (S_ - 1);
                int h  = gn >> 6,  hd = gn & (HD_ - 1);
                if (mode == 1) {          // [B,H,HD,S]: stride 1 in i -> pack
                    bf16* p = (bf16*)OutP + ((size_t)(bb * H_ + h) * HD_ + hd) * S_ + s0;
                    #pragma unroll
                    for (int i = 0; i < 8; i += 2)
                        store_bf2(&p[i], acc[ms][ns][i], acc[ms][ns][i + 1]);
                } else {                  // [B,H,S,HD]: stride HD_ in i
                    bf16* p = (bf16*)OutP + ((size_t)(bb * H_ + h) * S_ + s0) * HD_ + hd;
                    #pragma unroll
                    for (int i = 0; i < 8; ++i)
                        p[(size_t)i * HD_] = f2bf_fast(acc[ms][ns][i]);
                }
            }
        }
}

__global__ __launch_bounds__(256) void qkv_gemm(
    const bf16* __restrict__ Xb,
    const bf16* __restrict__ Wqt, const bf16* __restrict__ Wkt,
    const bf16* __restrict__ Wvt,
    bf16* __restrict__ Qo, bf16* __restrict__ Ko, bf16* __restrict__ Vo) {
    __shared__ __align__(16) GemmSmem sm;
    const bf16* Bt;  bf16* Out;  int mode;
    if (blockIdx.z == 0)      { Bt = Wqt; Out = Qo; mode = 0; }
    else if (blockIdx.z == 1) { Bt = Wkt; Out = Ko; mode = 0; }
    else                      { Bt = Wvt; Out = Vo; mode = 1; }
    gemm_core(sm, Xb, Bt, Out, mode);
}

__global__ __launch_bounds__(256) void out_proj(
    const bf16* __restrict__ A, const bf16* __restrict__ Wot,
    float* __restrict__ Out) {
    __shared__ __align__(16) GemmSmem sm;
    gemm_core(sm, A, Wot, Out, 2);
}

// ---------------------------------------------------------------------------
// Kernel 2: causal flash attention.  One (b, h, 64-query tile) per 128-thread
// block (4 waves x 16 query rows).  Double-buffered, symmetric TDM staging:
// waves 0-1 stage K halves, waves 2-3 stage V halves (pre-transposed V).
// ---------------------------------------------------------------------------
struct FlashSmem {
    bf16 Kl[2][64][72];   // [key][d], 9216 B per buffer
    bf16 Vl[2][64][72];   // [d][key]
    bf16 Pl[4][16][40];   // per-wave P scratch
};
#define KL_OFF(buf) ((unsigned)(buf) * 9216u)
#define VL_OFF(buf) (18432u + (unsigned)(buf) * 9216u)

__device__ __forceinline__ void flash_stage(FlashSmem& sm, int buf,
                                            const bf16* Kg, const bf16* Vg,
                                            int tid, int wave) {
#ifdef HAVE_TDM
    const bool isK = wave < 2;
    const bf16* src = isK ? (Kg + (size_t)(wave & 1) * 32 * HD_)
                          : (Vg + (size_t)(wave & 1) * 32 * S_);
    unsigned lds = (isK ? KL_OFF(buf) : VL_OFF(buf)) + (unsigned)(wave & 1) * 32u * 144u;
    unsigned d0  = isK ? HD_ : S_;       // contiguous extent == row pitch
    unsigned d1  = isK ? S_  : HD_;
    tdm_load_2d(lds, src, d0, d1, 64, 32, d0, 4);
#else
    #pragma unroll
    for (int p = 0; p < 4; ++p) {
        int row = p * 16 + (tid >> 3);
        int col = (tid & 7) * 8;
        *(bf16x8*)&sm.Kl[buf][row][col] = *(const bf16x8*)&Kg[(size_t)row * HD_ + col];
        *(bf16x8*)&sm.Vl[buf][row][col] = *(const bf16x8*)&Vg[(size_t)row * S_ + col];
    }
#endif
}

__global__ __launch_bounds__(128) void flash_attn(
    const bf16* __restrict__ Q, const bf16* __restrict__ Kd,
    const bf16* __restrict__ Vt, bf16* __restrict__ Oa) {

    __shared__ __align__(16) FlashSmem sm;

    const int tid  = threadIdx.x;
    const int wave = tid >> 5, lane = tid & 31, half = lane >> 4, r = lane & 15;
    const int qtile = blockIdx.x, h = blockIdx.y, b = blockIdx.z;
    const size_t hb = ((size_t)b * H_ + h) * S_ * HD_;
    const int q0 = qtile * 64 + wave * 16;

    bf16x16 qa[2];
    const bf16* qrow = Q + hb + (size_t)(q0 + r) * HD_;
    #pragma unroll
    for (int dc = 0; dc < 2; ++dc)
        qa[dc] = cat(*(const bf16x8*)&qrow[dc * 32 + half * 8],
                     *(const bf16x8*)&qrow[dc * 32 + 16 + half * 8]);

    float m[8], l[8];
    f32x8 o[4] = {};
    #pragma unroll
    for (int i = 0; i < 8; ++i) { m[i] = -3.0e38f; l[i] = 0.f; }

    flash_stage(sm, 0, Kd + hb, Vt + hb, tid, wave);          // prologue (kt=0)
    for (int kt = 0; kt <= qtile; ++kt) {
        if (kt + 1 <= qtile) {
            flash_stage(sm, (kt + 1) & 1,
                        Kd + hb + (size_t)(kt + 1) * 64 * HD_,
                        Vt + hb + (size_t)(kt + 1) * 64, tid, wave);
#ifdef HAVE_TDM
            wait_tensorcnt1();
#endif
        } else {
#ifdef HAVE_TDM
            wait_tensorcnt0();
#endif
        }
#ifdef HAVE_TDM
        lds_acquire(&sm);
#endif
        __syncthreads();
        const int bufi = kt & 1;

        #pragma unroll
        for (int kc = 0; kc < 2; ++kc) {
            const int c0 = kt * 64 + kc * 32;
            if (c0 > q0 + 15) continue;              // chunk fully above diagonal
            const bool needMask = (c0 + 31 > q0);

            f32x8 s0 = {}, s1 = {};
            #pragma unroll
            for (int ds = 0; ds < 2; ++ds) {
                int d0 = ds * 32 + half * 8;
                bf16x16 kb0 = cat(*(const bf16x8*)&sm.Kl[bufi][kc * 32 + r][d0],
                                  *(const bf16x8*)&sm.Kl[bufi][kc * 32 + r][d0 + 16]);
                bf16x16 kb1 = cat(*(const bf16x8*)&sm.Kl[bufi][kc * 32 + 16 + r][d0],
                                  *(const bf16x8*)&sm.Kl[bufi][kc * 32 + 16 + r][d0 + 16]);
                s0 = wmma_bf16(qa[ds], kb0, s0);
                s1 = wmma_bf16(qa[ds], kb1, s1);
            }

            float sc[8];
            #pragma unroll
            for (int i = 0; i < 8; ++i) {
                float f0 = s0[i] * 0.125f;           // 1/sqrt(64)
                float f1 = s1[i] * 0.125f;
                if (needMask) {
                    int qi = q0 + half * 8 + i;
                    if (c0 + r > qi)      f0 = -3.0e38f;
                    if (c0 + 16 + r > qi) f1 = -3.0e38f;
                }
                float cur = fmaxf(f0, f1);
                #pragma unroll
                for (int off = 1; off < 16; off <<= 1)
                    cur = fmaxf(cur, __shfl_xor(cur, off, 32));
                float mn = fmaxf(m[i], cur);
                float scale = __expf(m[i] - mn);
                m[i] = mn;
                float p0 = __expf(f0 - mn);
                float p1 = __expf(f1 - mn);
                float rsum = p0 + p1;
                #pragma unroll
                for (int off = 1; off < 16; off <<= 1)
                    rsum += __shfl_xor(rsum, off, 32);
                l[i] = l[i] * scale + rsum;
                sc[i] = scale;
                sm.Pl[wave][half * 8 + i][r]      = f2bf_fast(p0);
                sm.Pl[wave][half * 8 + i][16 + r] = f2bf_fast(p1);
            }
            __builtin_amdgcn_wave_barrier();         // keep ds stores before loads

            #pragma unroll
            for (int j = 0; j < 4; ++j)
                #pragma unroll
                for (int i = 0; i < 8; ++i)
                    o[j][i] *= sc[i];

            bf16x16 pf = cat(*(const bf16x8*)&sm.Pl[wave][r][half * 8],
                             *(const bf16x8*)&sm.Pl[wave][r][16 + half * 8]);

            #pragma unroll
            for (int j = 0; j < 4; ++j) {
                int d  = j * 16 + r;
                int kb = kc * 32 + half * 8;
                bf16x16 vb = cat(*(const bf16x8*)&sm.Vl[bufi][d][kb],
                                 *(const bf16x8*)&sm.Vl[bufi][d][kb + 16]);
                o[j] = wmma_bf16(pf, vb, o[j]);
            }
        }
        __syncthreads();
    }

    float rl[8];
    #pragma unroll
    for (int i = 0; i < 8; ++i) rl[i] = 1.0f / l[i];
    bf16* po = Oa + ((size_t)b * S_ + q0 + half * 8) * D_ + h * HD_ + r;
    #pragma unroll
    for (int j = 0; j < 4; ++j)
        #pragma unroll
        for (int i = 0; i < 8; ++i)
            po[(size_t)i * D_ + j * 16] = f2bf_fast(o[j][i] * rl[i]);
}

// ---------------------------------------------------------------------------
extern "C" void kernel_launch(void* const* d_in, const int* in_sizes, int n_in,
                              void* d_out, int out_size, void* d_ws, size_t ws_size,
                              hipStream_t stream) {
    const float* X  = (const float*)d_in[0];
    const float* Wq = (const float*)d_in[1];
    const float* Wk = (const float*)d_in[2];
    const float* Wv = (const float*)d_in[3];
    const float* Wo = (const float*)d_in[4];

    const size_t E  = (size_t)B_ * S_ * D_;    // 8,388,608
    const size_t WE = (size_t)D_ * D_;         // 1,048,576
    bf16* Xb  = (bf16*)d_ws;                   // bf16 X  [M,1024]
    bf16* Wqt = Xb  + E;                       // bf16 Wq^T [n][k]
    bf16* Wkt = Wqt + WE;
    bf16* Wvt = Wkt + WE;
    bf16* Wot = Wvt + WE;
    bf16* Qw  = Wot + WE;                      // bf16 Q  [B,H,S,HD]
    bf16* Kw  = Qw  + E;                       // bf16 K  [B,H,S,HD]
    bf16* Vw  = Kw  + E;                       // bf16 V  [B,H,HD,S] (transposed)
    bf16* Aw  = Vw  + E;                       // bf16 attn-out [B,S,D]

    convert_x<<<dim3((unsigned)(E / (8 * 256))), 256, 0, stream>>>(X, Xb);
    convert_wt<<<dim3(16, 16, 4), 256, 0, stream>>>(Wq, Wk, Wv, Wo,
                                                    Wqt, Wkt, Wvt, Wot);
    qkv_gemm<<<dim3(D_ / 128, M_TOT / 128, 3), 256, 0, stream>>>(
        Xb, Wqt, Wkt, Wvt, Qw, Kw, Vw);
    flash_attn<<<dim3(S_ / 64, H_, B_), 128, 0, stream>>>(Qw, Kw, Vw, Aw);
    out_proj<<<dim3(D_ / 128, M_TOT / 128, 1), 256, 0, stream>>>(Aw, Wot,
                                                                 (float*)d_out);
}
